// MotionAwareInterpolation_59502476919303
// MI455X (gfx1250) — matrix operand order repaired and checked by reading, not compiled, not run on hardware
//
#include <hip/hip_runtime.h>
#include <hip/hip_bf16.h>

typedef __attribute__((ext_vector_type(16))) _Float16 v16h;
typedef __attribute__((ext_vector_type(8)))  _Float16 v8h;
typedef __attribute__((ext_vector_type(8)))  float    v8f;

#define TILE 16
#define PD   24          // padded plane dim = TILE + 2*4 halo
#define HH   720
#define WW   1280
#define NB   4

// ---- LDS layout (byte offsets); all 32B-aligned ----
// All planes channel-padded to a multiple of 16 so one lane's 16 B-matrix
// elements (K = 16*half .. +15) are 16 consecutive f16 at a single (r,s) tap
// -> one 32B LDS read per WMMA operand.
#define OFF_IN   0
#define SZ_IN    (PD*PD*16*2)       // 18432  input, 6 real + 10 zero channels
#define OFF_P1   (OFF_IN + SZ_IN)
#define SZ_P1    (PD*PD*32*2)       // 36864  layer1 out
#define OFF_P2   (OFF_P1 + SZ_P1)
#define SZ_P2    (PD*PD*48*2)       // 55296  layer2 out
#define OFF_P3   (OFF_P2 + SZ_P2)
#define SZ_P3    (PD*PD*32*2)       // 36864  layer3 out
#define OFF_A1   (OFF_P3 + SZ_P3)
#define SZ_A1    (2*5*1024)         // MT x NK x 1KB per 16x32 f16 A-block
#define OFF_A2   (OFF_A1 + SZ_A1)
#define SZ_A2    (3*9*1024)
#define OFF_A3   (OFF_A2 + SZ_A2)
#define SZ_A3    (2*14*1024)
#define OFF_A4   (OFF_A3 + SZ_A3)
#define SZ_A4    (1*9*1024)
#define OFF_BIAS (OFF_A4 + SZ_A4)
#define SZ_BIAS  (128*4)
#define OFF_ZERO (OFF_BIAS + SZ_BIAS)   // 64B of zeros: K-tail redirect target
#define SZ_ZERO  64
#define SMEM_BYTES (OFF_ZERO + SZ_ZERO) // 223808 B < 320KB WGP LDS

// Rearrange f32 [O,I,3,3] weights into a 16-bit A-matrix lane layout with contiguous K
// (k = 16*half + element, not the ISA table order).  K ordering = (r*3+s)*CPAD + c, matching the channel-padded,
// channel-interleaved activation planes.  Zero outside (o<COUT, rs<9, c<CIN).
template<int CIN, int CPAD, int COUT, int NK>
__device__ __forceinline__ void prep_weights(const float* __restrict__ w,
                                             _Float16* __restrict__ A, int tid)
{
  const int MT  = (COUT + 15) / 16;
  const int TOT = MT * NK * 512;
  for (int i = tid; i < TOT; i += 256) {
    int blk  = i >> 9;
    int r    = i & 511;
    int lane = r >> 4;
    int v    = (r >> 1) & 7;
    int e    = r & 1;
    int mt   = blk / NK;
    int ks   = blk - mt * NK;
    int m    = lane & 15;
    int hf   = lane >> 4;
    int k0   = 16 * hf + 2 * v + e;
    int kg   = ks * 32 + k0;
    int o    = mt * 16 + m;
    int rs   = kg / CPAD;
    int c    = kg - rs * CPAD;
    _Float16 val = (_Float16)0.0f;
    if (o < COUT && rs < 9 && c < CIN) {
      int wr = rs / 3;
      int ws = rs - wr * 3;
      val = (_Float16)w[((o * CIN + c) * 3 + wr) * 3 + ws];
    }
    A[i] = val;   // i == blk*512 + lane*16 + v*2 + e
  }
}

// One conv layer as implicit GEMM with v_wmma_f32_16x16x32_f16.
// Layer L computes rows L..23-L at column strips x0 in {L, 8-L}.
// Per K-step: one 32B A read + one contiguous 32B B read + one WMMA.
template<int CPAD, int COUT, int NK, int L>
__device__ __forceinline__ void conv_layer(const char* smem,
    int pin_off, int pout_off, int a_off, int bias_off,
    int wid, int lane, int n, int hf)
{
  const _Float16* pin  = (const _Float16*)(smem + pin_off);
  _Float16*       pout = (_Float16*)(smem + pout_off);
  const _Float16* A    = (const _Float16*)(smem + a_off);
  const float*    bias = (const float*)(smem + bias_off);
  const _Float16* zp   = (const _Float16*)(smem + OFF_ZERO);
  const int MT    = COUT / 16;
  const int NROWS = PD - 2 * L;
  const int NT    = NROWS * 2 * MT;
  const int KTOT  = 9 * CPAD;
  for (int t = wid; t < NT; t += 8) {          // wave-uniform: EXEC full for WMMA
    int mt    = t % MT;
    int rest  = t / MT;
    int strip = rest & 1;
    int row   = L + (rest >> 1);
    int x0    = strip ? (8 - L) : L;
    const _Float16* Ab = A + (mt * NK) * 512 + lane * 16;
    v8f acc = {0.f, 0.f, 0.f, 0.f, 0.f, 0.f, 0.f, 0.f};
    for (int ks = 0; ks < NK; ++ks) {
      int kg = ks * 32 + hf * 16;              // first K this lane-half covers
      int rs = kg / CPAD;                      // shift for 16/32, magic-mul for 48
      int c0 = kg - rs * CPAD;
      int wr = rs / 3;
      int ws = rs - wr * 3;
      int y  = row + wr - 1;
      int x  = x0 + n + ws - 1;
      const _Float16* bp = (kg < KTOT) ? (pin + (y * PD + x) * CPAD + c0) : zp;
      v16h bv = *(const v16h*)bp;              // 2x ds_load_b128, 16 consecutive ch
      v16h av = *(const v16h*)(Ab + ks * 512); // 2x ds_load_b128
      acc = __builtin_amdgcn_wmma_f32_16x16x32_f16(false, av, false, bv,
                                                   (short)0, acc, false, false);
    }
    // D lane layout: 8 accumulators = 8 consecutive output channels -> one b128 store
    int base = mt * 16 + 8 * hf;
    v8h dh;
#pragma unroll
    for (int v = 0; v < 8; ++v) {
      float r = acc[v] + bias[base + v];
      dh[v] = (_Float16)fmaxf(r, 0.0f);        // ReLU (layers 1-3)
    }
    *(v8h*)(pout + (row * PD + x0 + n) * COUT + base) = dh;
  }
}

__device__ __forceinline__ float sigmoidf(float z) {
  return 1.0f / (1.0f + __expf(-z));
}

// Layer 4 (32->3, CPAD=32) + smooth-mask blend + clip, straight to global f32.
__device__ __forceinline__ void conv_layer4_out(const char* smem,
    const float* __restrict__ f1, const float* __restrict__ f3,
    float* __restrict__ out,
    int b, int gy0, int gx0, int wid, int lane, int n, int hf)
{
  const _Float16* pin = (const _Float16*)(smem + OFF_P3);
  const _Float16* A   = (const _Float16*)(smem + OFF_A4);
  const float*    b4  = (const float*)(smem + OFF_BIAS) + 112;
  for (int t = wid; t < 16; t += 8) {
    int row = 4 + t;
    const int x0 = 4;
    const _Float16* Ab = A + lane * 16;
    v8f acc = {0.f, 0.f, 0.f, 0.f, 0.f, 0.f, 0.f, 0.f};
    for (int ks = 0; ks < 9; ++ks) {           // 288 = 9*32 exact: no tail guard
      int kg = ks * 32 + hf * 16;
      int rs = kg >> 5;
      int c0 = kg & 31;
      int wr = rs / 3;
      int ws = rs - wr * 3;
      int y  = row + wr - 1;
      int x  = x0 + n + ws - 1;
      v16h bv = *(const v16h*)(pin + (y * PD + x) * 32 + c0);
      v16h av = *(const v16h*)(Ab + ks * 512);
      acc = __builtin_amdgcn_wmma_f32_16x16x32_f16(false, av, false, bv,
                                                   (short)0, acc, false, false);
    }
    if (hf == 0) {                             // channels 0..2 live in acc[0..2], lanes 0-15
      int gy = gy0 + row - 4;
      int gx = gx0 + n;
      float tx = (float)gx * (1.0f / (float)(WW - 1));
      float ty = (float)gy * (1.0f / (float)(HH - 1));
      float mx = fminf(sigmoidf((tx - 0.1f) * 10.0f), sigmoidf((0.9f - tx) * 10.0f));
      float my = fminf(sigmoidf((ty - 0.1f) * 10.0f), sigmoidf((0.9f - ty) * 10.0f));
      float m  = mx * my;
#pragma unroll
      for (int ch = 0; ch < 3; ++ch) {
        size_t idx = ((size_t)(b * 3 + ch) * HH + gy) * WW + gx;
        float region = acc[ch] + b4[ch];
        float avg    = 0.5f * (f1[idx] + f3[idx]);     // f32 re-read for exact avg
        float o      = avg * (1.0f - m) + region * m;
        out[idx] = fminf(fmaxf(o, -1.0f), 1.0f);
      }
    }
  }
}

__global__ void __launch_bounds__(256, 1)
motion_interp_fused(const float* __restrict__ f1, const float* __restrict__ f3,
                    const float* __restrict__ w1, const float* __restrict__ b1,
                    const float* __restrict__ w2, const float* __restrict__ b2,
                    const float* __restrict__ w3, const float* __restrict__ b3,
                    const float* __restrict__ w4, const float* __restrict__ b4,
                    float* __restrict__ out)
{
  extern __shared__ char smem[];
  int tid  = threadIdx.x;
  int lane = tid & 31;
  int wid  = tid >> 5;
  int n    = lane & 15;   // N (pixel) index within WMMA tile
  int hf   = lane >> 4;   // half-wave selector for K/M striping
  int gx0  = blockIdx.x * TILE;
  int gy0  = blockIdx.y * TILE;
  int b    = blockIdx.z;

  // ---- phase 0: stage input halo tile (f32->f16, 16-ch interleaved, one 32B
  //      store per pixel), pre-swizzled weights, bias, zero block
  {
    _Float16* pin = (_Float16*)(smem + OFF_IN);
    for (int i = tid; i < PD * PD; i += 256) {
      int x  = i % PD;
      int y  = i / PD;
      int gy = gy0 - 4 + y;
      int gx = gx0 - 4 + x;
      bool inb = (gy >= 0 && gy < HH && gx >= 0 && gx < WW);
      size_t gidx = ((size_t)(b * 3) * HH + gy) * WW + gx;
      v16h px = {};                           // channels 6..15 stay zero
#pragma unroll
      for (int c = 0; c < 3; ++c) {
        px[c]     = inb ? (_Float16)f1[gidx + (size_t)c * HH * WW] : (_Float16)0.0f;
        px[c + 3] = inb ? (_Float16)f3[gidx + (size_t)c * HH * WW] : (_Float16)0.0f;
      }
      *(v16h*)(pin + i * 16) = px;
    }
  }
  prep_weights<6,  16, 32, 5 >(w1, (_Float16*)(smem + OFF_A1), tid);
  prep_weights<32, 32, 48, 9 >(w2, (_Float16*)(smem + OFF_A2), tid);
  prep_weights<48, 48, 32, 14>(w3, (_Float16*)(smem + OFF_A3), tid);
  prep_weights<32, 32, 3,  9 >(w4, (_Float16*)(smem + OFF_A4), tid);
  {
    float* bias = (float*)(smem + OFF_BIAS);
    if (tid < 32) bias[tid]       = b1[tid];
    if (tid < 48) bias[32 + tid]  = b2[tid];
    if (tid < 32) bias[80 + tid]  = b3[tid];
    if (tid < 3)  bias[112 + tid] = b4[tid];
    if (tid < 16) ((unsigned int*)(smem + OFF_ZERO))[tid] = 0u;
  }
  __syncthreads();

  conv_layer<16, 32, 5,  1>(smem, OFF_IN, OFF_P1, OFF_A1, OFF_BIAS + 0,      wid, lane, n, hf);
  __syncthreads();
  conv_layer<32, 48, 9,  2>(smem, OFF_P1, OFF_P2, OFF_A2, OFF_BIAS + 32 * 4, wid, lane, n, hf);
  __syncthreads();
  conv_layer<48, 32, 14, 3>(smem, OFF_P2, OFF_P3, OFF_A3, OFF_BIAS + 80 * 4, wid, lane, n, hf);
  __syncthreads();
  conv_layer4_out(smem, f1, f3, out, b, gy0, gx0, wid, lane, n, hf);
}

extern "C" void kernel_launch(void* const* d_in, const int* in_sizes, int n_in,
                              void* d_out, int out_size, void* d_ws, size_t ws_size,
                              hipStream_t stream) {
  (void)in_sizes; (void)n_in; (void)out_size; (void)d_ws; (void)ws_size;
  const float* f1 = (const float*)d_in[0];
  const float* f3 = (const float*)d_in[1];
  const float* w1 = (const float*)d_in[2];
  const float* b1 = (const float*)d_in[3];
  const float* w2 = (const float*)d_in[4];
  const float* b2 = (const float*)d_in[5];
  const float* w3 = (const float*)d_in[6];
  const float* b3 = (const float*)d_in[7];
  const float* w4 = (const float*)d_in[8];
  const float* b4 = (const float*)d_in[9];
  float* out = (float*)d_out;

  // Opt-in to >64KB dynamic LDS (harmless if not required on this stack).
  (void)hipFuncSetAttribute((const void*)motion_interp_fused,
                            hipFuncAttributeMaxDynamicSharedMemorySize,
                            (int)SMEM_BYTES);

  dim3 grid(WW / TILE, HH / TILE, NB);   // 80 x 45 x 4
  dim3 block(256);                        // 8 wave32 per workgroup
  motion_interp_fused<<<grid, block, SMEM_BYTES, stream>>>(
      f1, f3, w1, b1, w2, b2, w3, b3, w4, b4, out);
}
